// MultiHeadAttention_7438883356839
// MI455X (gfx1250) — matrix operand order repaired and checked by reading, not compiled, not run on hardware
//
#include <hip/hip_runtime.h>
#include <hip/hip_bf16.h>

// ---------------------------------------------------------------------------
// CDNA5 (gfx1250) multi-head attention, bf16 WMMA pipeline:
//   stage 0: one-shot f32->bf16 conversion of X and W (bandwidth pass)
//   stage 1: Q/K = X @ W^T + b -> bf16 ws ;  V additionally stored V^T/head
//   stage 2: flash attention, one wave32 per (b,h,16-row q tile)
//   stage 3: out = Ctx @ Wo^T + bo -> f32 d_out
// B=2, S=2048, D=1024, H=16, HD=64 (shifts hardcoded where noted).
// ---------------------------------------------------------------------------

typedef __attribute__((ext_vector_type(16))) __bf16          v16bf;
typedef __attribute__((ext_vector_type(8)))  float           v8f;
typedef __attribute__((ext_vector_type(8)))  unsigned short  v8us;

union FragBF { v16bf bf; unsigned short us[16]; };

__device__ __forceinline__ unsigned short bfbits(float f) {
  union { __bf16 h; unsigned short u; } cv;
  cv.h = (__bf16)f;
  return cv.u;
}

__device__ __forceinline__ void load8bf(const unsigned short* __restrict__ p,
                                        unsigned short* d) {
  v8us t = *(const v8us*)p;               // 16-byte load
#pragma unroll
  for (int i = 0; i < 8; ++i) d[i] = t[i];
}

__device__ __forceinline__ v8f wmma_bf16(const FragBF& a, const FragBF& b, v8f c) {
  return __builtin_amdgcn_wmma_f32_16x16x32_bf16(
      false, a.bf, false, b.bf, (short)0, c, false, false);
}

// xor-butterfly within each 16-lane row, pure VALU via v_permlane16_b32.
__device__ __forceinline__ float permxor16(float v, int step,
                                           unsigned lo, unsigned hi) {
#if __has_builtin(__builtin_amdgcn_permlane16)
  unsigned u = __float_as_uint(v);
  u = __builtin_amdgcn_permlane16(u, u, lo, hi, false, false);
  return __uint_as_float(u);
#else
  (void)lo; (void)hi;
  return __shfl_xor(v, step, 32);
#endif
}

__constant__ const unsigned kSelLo[4] = {0x67452301u, 0x54761302u,
                                         0x32107654u, 0xFEDCBA98u};
__constant__ const unsigned kSelHi[4] = {0xEFCDAB89u, 0xDCFE98BAu,
                                         0xBA98FEDCu, 0x76543210u};

// ---------------------------------------------------------------------------
// Stage 0: bulk f32 -> bf16 (8 elements/lane, 32B in -> 16B out)
// ---------------------------------------------------------------------------
__global__ __launch_bounds__(256)
void cvt_bf16_kernel(const float* __restrict__ in,
                     unsigned short* __restrict__ out) {
  const size_t i = ((size_t)blockIdx.x * 256 + threadIdx.x) * 8;
  const float4 v0 = *(const float4*)(in + i);
  const float4 v1 = *(const float4*)(in + i + 4);
  union { unsigned short us[8]; v8us v; } o;
  o.us[0] = bfbits(v0.x); o.us[1] = bfbits(v0.y);
  o.us[2] = bfbits(v0.z); o.us[3] = bfbits(v0.w);
  o.us[4] = bfbits(v1.x); o.us[5] = bfbits(v1.y);
  o.us[6] = bfbits(v1.z); o.us[7] = bfbits(v1.w);
  *(v8us*)(out + i) = o.v;
}

// ---------------------------------------------------------------------------
// Stage 1: C[M,N] = A[M,K] @ W[N,K]^T + bias   (bf16 in, bf16 out)
// One wave per 32x32 macro-tile: 2x2 fragments, 4 WMMAs per K-step,
// conversion-free inner loop (pure b128 loads + WMMA).
// VT=true stores transposed per head: Out[((b*16+h)*64+d)*2048 + s].
// ---------------------------------------------------------------------------
template <bool VT>
__global__ __launch_bounds__(128)
void gemm_bf16_kernel(const unsigned short* __restrict__ A,
                      const unsigned short* __restrict__ W,
                      const float* __restrict__ bias,
                      unsigned short* __restrict__ Out,
                      int M, int K, int N) {
  const int lane = threadIdx.x & 31;
  const int hl   = lane >> 4;
  const int l15  = lane & 15;
  const int tile = blockIdx.x * 4 + (threadIdx.x >> 5);
  const int ntn  = N >> 5;
  const int tn   = (tile % ntn) << 5;
  const int tm   = (tile / ntn) << 5;
  if (tm >= M) return;

  const unsigned short* a0p = A + (size_t)(tm + l15) * K + (hl << 3);
  const unsigned short* a1p = a0p + (size_t)16 * K;
  const unsigned short* b0p = W + (size_t)(tn + l15) * K + (hl << 3);
  const unsigned short* b1p = b0p + (size_t)16 * K;

  v8f c00 = {}, c01 = {}, c10 = {}, c11 = {};
  for (int kb = 0; kb < K; kb += 32) {
    FragBF a0, a1, b0, b1;
    load8bf(a0p +  0, a0.us + 0);
    load8bf(a0p + 16, a0.us + 8);
    load8bf(a1p +  0, a1.us + 0);
    load8bf(a1p + 16, a1.us + 8);
    load8bf(b0p +  0, b0.us + 0);
    load8bf(b0p + 16, b0.us + 8);
    load8bf(b1p +  0, b1.us + 0);
    load8bf(b1p + 16, b1.us + 8);
    a0p += 32; a1p += 32; b0p += 32; b1p += 32;
    c00 = wmma_bf16(a0, b0, c00);
    c01 = wmma_bf16(a0, b1, c01);
    c10 = wmma_bf16(a1, b0, c10);
    c11 = wmma_bf16(a1, b1, c11);
  }

  const float bn0 = bias[tn + l15];
  const float bn1 = bias[tn + 16 + l15];

  auto stor = [&](v8f c, int rbase, int cbase, float bn) {
#pragma unroll
    for (int r = 0; r < 8; ++r) {
      const int row = rbase + r + (hl << 3);
      const int col = cbase + l15;
      const unsigned short v = bfbits(c[r] + bn);
      if (VT) {                      // per-head transpose: S=2048, HD=64, H=16
        const int b = row >> 11, s = row & 2047;
        const int h = col >> 6,  d = col & 63;
        Out[(((size_t)((b << 4) + h) << 6) + d) * 2048 + s] = v;
      } else {
        Out[(size_t)row * N + col] = v;
      }
    }
  };
  stor(c00, tm,      tn,      bn0);
  stor(c01, tm,      tn + 16, bn1);
  stor(c10, tm + 16, tn,      bn0);
  stor(c11, tm + 16, tn + 16, bn1);
}

// ---------------------------------------------------------------------------
// Stage 2: flash attention.  One wave32 per (b, h, 16-row q tile).
// Q,K: [B*S, D] bf16 row-major.  Vt: [B][H][HD][S] bf16 (s contiguous).
// ---------------------------------------------------------------------------
__global__ __launch_bounds__(128)
void attn_kernel(const unsigned short* __restrict__ Q,
                 const unsigned short* __restrict__ Kp,
                 const unsigned short* __restrict__ Vt,
                 unsigned short* __restrict__ Ctx,
                 int S, int D) {
  __shared__ __align__(16) unsigned short Pld[4][16 * 32];

  const int wave = threadIdx.x >> 5;
  const int lane = threadIdx.x & 31;
  const int hl   = lane >> 4;
  const int l15  = lane & 15;

  const int id  = blockIdx.x * 4 + wave;
  const int qt  = id & 127;                   // S/16 = 128
  const int h   = (id >> 7) & 15;             // H  = 16
  const int b   = id >> 11;
  const int rowQ = b * S + qt * 16;
  const int col0 = h << 6;                    // HD = 64

  FragBF qa0, qa1;                            // Q A-fragments d=[0,32),[32,64)
  {
    const unsigned short* qrow = Q + (size_t)(rowQ + l15) * D + col0;
    const int ab = hl << 3;
    load8bf(qrow + ab,           qa0.us + 0);
    load8bf(qrow + ab + 16,      qa0.us + 8);
    load8bf(qrow + 32 + ab,      qa1.us + 0);
    load8bf(qrow + 32 + ab + 16, qa1.us + 8);
  }

  float mrow[8], lrow[8];
  v8f acc[4];
#pragma unroll
  for (int r = 0; r < 8; ++r) { mrow[r] = -3.0e38f; lrow[r] = 0.0f; }
#pragma unroll
  for (int t = 0; t < 4; ++t) { v8f z = {}; acc[t] = z; }

  unsigned short* P = Pld[wave];
  const unsigned short* vhead = Vt + ((size_t)((b << 4) + h) << 6) * S;
  const float sqk = 0.125f;                   // 1/sqrt(64)
  const int steps[4] = {1, 2, 4, 8};

  // Streaming pointers: constant-stride increments, base+imm-offset loads.
  const size_t strideK = (size_t)32 * D;
  const unsigned short* kb0 = Kp + (size_t)(b * S + l15) * D + col0 + (hl << 3);
  const unsigned short* kb1 = kb0 + (size_t)16 * D;
  const unsigned short* vp[4];
#pragma unroll
  for (int t = 0; t < 4; ++t)
    vp[t] = vhead + (size_t)(t * 16 + l15) * S + (hl << 3);

  for (int kk = 0; kk < S; kk += 32) {
    // ---- scores for 32 keys: two 16x16 tiles, contraction over d=64 -------
    v8f s0 = {}, s1 = {};
    {
      __builtin_prefetch(kb0 + strideK, 0, 1);
      FragBF k0a, k0b, k1a, k1b;
      load8bf(kb0 +  0, k0a.us + 0);
      load8bf(kb0 + 16, k0a.us + 8);
      load8bf(kb0 + 32, k0b.us + 0);
      load8bf(kb0 + 48, k0b.us + 8);
      load8bf(kb1 +  0, k1a.us + 0);
      load8bf(kb1 + 16, k1a.us + 8);
      load8bf(kb1 + 32, k1b.us + 0);
      load8bf(kb1 + 48, k1b.us + 8);
      s0 = wmma_bf16(qa0, k0a, s0);
      s0 = wmma_bf16(qa1, k0b, s0);
      s1 = wmma_bf16(qa0, k1a, s1);
      s1 = wmma_bf16(qa1, k1b, s1);
    }
    kb0 += strideK; kb1 += strideK;

    // ---- online softmax: row reductions via v_permlane16_b32 butterfly ----
    float av[8], cv[8], tmax[8], alpha[8], rs[8];
#pragma unroll
    for (int r = 0; r < 8; ++r) {
      av[r] = s0[r] * sqk;
      cv[r] = s1[r] * sqk;
      tmax[r] = fmaxf(av[r], cv[r]);
    }
#pragma unroll
    for (int st = 0; st < 4; ++st)
#pragma unroll
      for (int r = 0; r < 8; ++r)
        tmax[r] = fmaxf(tmax[r],
                        permxor16(tmax[r], steps[st], kSelLo[st], kSelHi[st]));
#pragma unroll
    for (int r = 0; r < 8; ++r) {
      const float m2 = fmaxf(mrow[r], tmax[r]);
      alpha[r] = __expf(mrow[r] - m2);
      mrow[r] = m2;
      const float p0 = __expf(av[r] - m2);
      const float p1 = __expf(cv[r] - m2);
      s0[r] = p0; s1[r] = p1;
      rs[r] = p0 + p1;
    }
#pragma unroll
    for (int st = 0; st < 4; ++st)
#pragma unroll
      for (int r = 0; r < 8; ++r)
        rs[r] += permxor16(rs[r], steps[st], kSelLo[st], kSelHi[st]);
#pragma unroll
    for (int r = 0; r < 8; ++r) {
      lrow[r] = lrow[r] * alpha[r] + rs[r];
#pragma unroll
      for (int t = 0; t < 4; ++t) acc[t][r] *= alpha[r];
    }

    // ---- transpose P: C layout -> A layout through per-wave LDS tile ------
#pragma unroll
    for (int r = 0; r < 8; ++r) {
      const int row = r + (hl << 3);
      P[row * 32 + l15]      = bfbits(s0[r]);
      P[row * 32 + 16 + l15] = bfbits(s1[r]);
    }
    asm volatile("s_wait_dscnt 0" ::: "memory"); // LDS RAW fence (same wave)

    FragBF pa;
    {
      const unsigned short* pr = P + l15 * 32;
      const int ab = hl << 3;
      load8bf(pr + ab,      pa.us + 0);
      load8bf(pr + ab + 16, pa.us + 8);
    }

    // ---- P(16x32) x V(32x16): Vt rows are s-contiguous -> b128 loads ------
#pragma unroll
    for (int t = 0; t < 4; ++t) {
      FragBF vb;
      load8bf(vp[t],     vb.us + 0);
      load8bf(vp[t] + 16, vb.us + 8);
      vp[t] += 32;
      acc[t] = wmma_bf16(pa, vb, acc[t]);
    }
  }

  // ---- normalize and store context (bf16, merged-head layout) -------------
#pragma unroll
  for (int r = 0; r < 8; ++r) {
    const int row = rowQ + r + (hl << 3);
    const float inv = 1.0f / lrow[r];
#pragma unroll
    for (int t = 0; t < 4; ++t)
      Ctx[(size_t)row * D + col0 + t * 16 + l15] = bfbits(acc[t][r] * inv);
  }
}

// ---------------------------------------------------------------------------
// Stage 3: Out[M,N] = Ctx[M,K](bf16) @ Wo[N,K]^T(bf16) + bo  (f32 out)
// ---------------------------------------------------------------------------
__global__ __launch_bounds__(128)
void out_proj_kernel(const unsigned short* __restrict__ A,
                     const unsigned short* __restrict__ W,
                     const float* __restrict__ bias,
                     float* __restrict__ Out,
                     int M, int K, int N) {
  const int lane = threadIdx.x & 31;
  const int hl   = lane >> 4;
  const int l15  = lane & 15;
  const int tile = blockIdx.x * 4 + (threadIdx.x >> 5);
  const int ntn  = N >> 5;
  const int tn   = (tile % ntn) << 5;
  const int tm   = (tile / ntn) << 5;
  if (tm >= M) return;

  const unsigned short* a0p = A + (size_t)(tm + l15) * K + (hl << 3);
  const unsigned short* a1p = a0p + (size_t)16 * K;
  const unsigned short* b0p = W + (size_t)(tn + l15) * K + (hl << 3);
  const unsigned short* b1p = b0p + (size_t)16 * K;

  v8f c00 = {}, c01 = {}, c10 = {}, c11 = {};
  for (int kb = 0; kb < K; kb += 32) {
    FragBF a0, a1, b0, b1;
    load8bf(a0p +  0, a0.us + 0);
    load8bf(a0p + 16, a0.us + 8);
    load8bf(a1p +  0, a1.us + 0);
    load8bf(a1p + 16, a1.us + 8);
    load8bf(b0p +  0, b0.us + 0);
    load8bf(b0p + 16, b0.us + 8);
    load8bf(b1p +  0, b1.us + 0);
    load8bf(b1p + 16, b1.us + 8);
    a0p += 32; a1p += 32; b0p += 32; b1p += 32;
    c00 = wmma_bf16(a0, b0, c00);
    c01 = wmma_bf16(a0, b1, c01);
    c10 = wmma_bf16(a1, b0, c10);
    c11 = wmma_bf16(a1, b1, c11);
  }

  const float bn0 = bias[tn + l15];
  const float bn1 = bias[tn + 16 + l15];
  auto stor = [&](v8f c, int rbase, int cbase, float bn) {
#pragma unroll
    for (int r = 0; r < 8; ++r)
      Out[(size_t)(rbase + r + (hl << 3)) * N + cbase + l15] = c[r] + bn;
  };
  stor(c00, tm,      tn,      bn0);
  stor(c01, tm,      tn + 16, bn1);
  stor(c10, tm + 16, tn,      bn0);
  stor(c11, tm + 16, tn + 16, bn1);
}

// ---------------------------------------------------------------------------
extern "C" void kernel_launch(void* const* d_in, const int* in_sizes, int n_in,
                              void* d_out, int out_size, void* d_ws, size_t ws_size,
                              hipStream_t stream) {
  const float* query = (const float*)d_in[0];
  const float* key_  = (const float*)d_in[1];
  const float* value = (const float*)d_in[2];
  const float* Wq    = (const float*)d_in[3];
  const float* bq    = (const float*)d_in[4];
  const float* Wk    = (const float*)d_in[5];
  const float* bk    = (const float*)d_in[6];
  const float* Wv    = (const float*)d_in[7];
  const float* bv    = (const float*)d_in[8];
  const float* Wo    = (const float*)d_in[9];
  const float* bo    = (const float*)d_in[10];

  const int Bn = 2, S = 2048, D = 1024;
  const int M = Bn * S;                             // 4096 rows
  const size_t MD = (size_t)M * D;                  // 4M elements
  const size_t DD = (size_t)D * D;                  // 1M elements

  // Workspace (bf16): Q | K | Vt | Ctx | Xb (reused) | Wb (reused)  = 42 MB
  unsigned short* Qb  = (unsigned short*)d_ws;
  unsigned short* Kb  = Qb + MD;
  unsigned short* Vtb = Kb + MD;
  unsigned short* Cb  = Vtb + MD;
  unsigned short* Xb  = Cb + MD;
  unsigned short* Wb  = Xb + MD;

  const dim3 blk(128);                              // 4 waves (wave32)
  const int projBlocks = (M / 32) * (D / 32) / 4;   // 1024
  const int attnBlocks = (M / 16) / 4;              // 64 (= B*S/16 / 4)
  const int cvtX = (int)(MD / (256 * 8));           // 2048
  const int cvtW = (int)(DD / (256 * 8));           // 512

  // Q projection
  cvt_bf16_kernel<<<cvtX, 256, 0, stream>>>(query, Xb);
  cvt_bf16_kernel<<<cvtW, 256, 0, stream>>>(Wq, Wb);
  gemm_bf16_kernel<false><<<projBlocks, blk, 0, stream>>>(Xb, Wb, bq, Qb, M, D, D);
  // K projection
  cvt_bf16_kernel<<<cvtX, 256, 0, stream>>>(key_, Xb);
  cvt_bf16_kernel<<<cvtW, 256, 0, stream>>>(Wk, Wb);
  gemm_bf16_kernel<false><<<projBlocks, blk, 0, stream>>>(Xb, Wb, bk, Kb, M, D, D);
  // V projection (stores V^T per head)
  cvt_bf16_kernel<<<cvtX, 256, 0, stream>>>(value, Xb);
  cvt_bf16_kernel<<<cvtW, 256, 0, stream>>>(Wv, Wb);
  gemm_bf16_kernel<true ><<<projBlocks, blk, 0, stream>>>(Xb, Wb, bv, Vtb, M, D, D);
  // Attention
  attn_kernel<<<attnBlocks, blk, 0, stream>>>(Qb, Kb, Vtb, Cb, S, D);
  // Output projection
  cvt_bf16_kernel<<<cvtW, 256, 0, stream>>>(Wo, Wb);
  out_proj_kernel<<<projBlocks, blk, 0, stream>>>(Cb, Wb, bo, (float*)d_out, M, D, D);
}
